// SwinTransformerBlock_83494164234929
// MI455X (gfx1250) — matrix operand order repaired and checked by reading, not compiled, not run on hardware
//
#include <hip/hip_runtime.h>
#include <stdint.h>

// ---------------------------------------------------------------------------
// Swin-V2 block for MI455X (gfx1250), wave32, bf16 WMMA 16x16x32, f32 accum.
// B operands staged to LDS with gfx1250 async global->LDS DMA (ASYNCcnt).
// ---------------------------------------------------------------------------

typedef __bf16 bf16;
typedef __attribute__((ext_vector_type(16))) __bf16 v16bf;
typedef __attribute__((ext_vector_type(8)))  __bf16 v8bf;
typedef __attribute__((ext_vector_type(8)))  float  v8f;

#define DIMC   384
#define NHEAD  12
#define HDIM   32
#define NTOK   64
#define HIDF   1536
#define MTOT   131072      // 2048 windows * 64 tokens
#define LDQKV  1152
#define BSTR   40          // padded K-stride (bf16) of LDS B rows (80B, 16B-aligned)

// ---- workspace byte offsets (aliasing is safe by stream ordering) ----------
#define OFF_WQKV   0u               // qkv_w^T  [1152][384] bf16
#define OFF_WPROJ  884736u          // proj_w^T [384][384]  bf16
#define OFF_WFC1   1179648u         // fc1_w^T  [1536][384] bf16
#define OFF_WFC2   2359296u         // fc2_w^T  [384][1536] bf16
#define OFF_QKVB   3538944u         // 1152 f32 (q_bias, 0, v_bias)
#define OFF_RPB    3543552u         // 12*64*64 f32 (16*sigmoid(cpb))
#define OFF_REGION 3740160u         // 64*64 i32 region ids
#define OFF_HSC    3756544u         // 12 f32 head scales
#define OFF_XW     4194304u         // MTOT*384 bf16 (reused as ATTOUT)
#define OFF_QKV    104857600u       // MTOT*1152 bf16 (V third unused)
#define OFF_XMIDF  104857600u       // alias: MTOT*384 f32 residual stream
#define OFF_XMIDB  306184192u       // first VT (transposed V), then XMID bf16
#define OFF_VT     306184192u       // [2048*12][32][64] bf16

__device__ __forceinline__ v8f zero8() {
  v8f z;
#pragma unroll
  for (int e = 0; e < 8; ++e) z[e] = 0.0f;
  return z;
}

__device__ __forceinline__ v16bf ld_pair16(const bf16* p0, const bf16* p1) {
  union { v16bf v; v8bf h[2]; } u;
  u.h[0] = *(const v8bf*)p0;
  u.h[1] = *(const v8bf*)p1;
  return u.v;
}

__device__ __forceinline__ v8f wmma_bf16(v16bf a, v16bf b, v8f c) {
  return __builtin_amdgcn_wmma_f32_16x16x32_bf16(false, a, false, b, (short)0, c,
                                                 false, false);
}

// A fragment (16x32 bf16) from row-major storage.
__device__ __forceinline__ v16bf afrag(const bf16* A, int lda, int row0,
                                       int kbase, int lane) {
  const int r  = row0 + (lane & 15);
  const int kc = kbase + ((lane >> 4) << 3);
  const bf16* p = A + (long)r * lda + kc;
  return ld_pair16(p, p + 16);
}

// B fragment (32x16 bf16) from K-contiguous storage with row stride ldk.
__device__ __forceinline__ v16bf bfrag(const bf16* Bt, int ldk, int n0,
                                       int kbase, int lane) {
  const int n  = n0 + (lane & 15);
  const int kc = kbase + ((lane >> 4) << 3);
  const bf16* p = Bt + (long)n * ldk + kc;
  return ld_pair16(p, p + 16);
}

// ---- gfx1250 async global->LDS copy (per-lane 16B), ASYNCcnt tracked -------
__device__ __forceinline__ void async_copy_b128(unsigned lds_off,
                                                const void* gptr) {
  asm volatile("global_load_async_to_lds_b128 %0, %1, off"
               :: "v"(lds_off), "v"((unsigned long long)(uintptr_t)gptr)
               : "memory");
}
__device__ __forceinline__ void async_wait0() {
  asm volatile("s_wait_asynccnt 0" ::: "memory");
}

// Stage rows [n0, n0+NCOL) x K-chunk [kbase, kbase+32) of WT (K-contiguous,
// row stride ldk) into LDS rows bl[n][BSTR] via async DMA. blockDim must = 256.
template <int NCOL>
__device__ __forceinline__ void stage_async(const bf16* WT, int ldk, int n0,
                                            int kbase, unsigned bl_off,
                                            int tid) {
#pragma unroll
  for (int c = tid; c < NCOL * 4; c += 256) {
    const int n = c >> 2, pc = c & 3;
    const bf16* g = WT + (long)(n0 + n) * ldk + kbase + pc * 8;
    async_copy_b128(bl_off + (unsigned)(n * (BSTR * 2) + pc * 16), g);
  }
}

// ---------------------------------------------------------------------------
// Setup: weight f32 -> bf16 TRANSPOSED (K-contiguous) + fused qkv bias
// ---------------------------------------------------------------------------
__global__ __launch_bounds__(256) void k_convert(
    const float* qkv_w, const float* proj_w, const float* fc1_w,
    const float* fc2_w, const float* q_bias, const float* v_bias, char* ws) {
  bf16* wq = (bf16*)(ws + OFF_WQKV);
  bf16* wp = (bf16*)(ws + OFF_WPROJ);
  bf16* w1 = (bf16*)(ws + OFF_WFC1);
  bf16* w2 = (bf16*)(ws + OFF_WFC2);
  float* qb = (float*)(ws + OFF_QKVB);
  unsigned g = blockIdx.x * 256u + threadIdx.x;
  if (g < 442368u) {                       // wq^T[n=0..1151][k=0..383]
    unsigned n = g / 384u, k = g - n * 384u;
    wq[g] = (bf16)qkv_w[k * 1152u + n];
  } else if (g < 589824u) {                // wp^T[n=0..383][k=0..383]
    unsigned i = g - 442368u, n = i / 384u, k = i - n * 384u;
    wp[i] = (bf16)proj_w[k * 384u + n];
  } else if (g < 1179648u) {               // w1^T[n=0..1535][k=0..383]
    unsigned i = g - 589824u, n = i / 384u, k = i - n * 384u;
    w1[i] = (bf16)fc1_w[k * 1536u + n];
  } else if (g < 1769472u) {               // w2^T[n=0..383][k=0..1535]
    unsigned i = g - 1179648u, n = i / 1536u, k = i - n * 1536u;
    w2[i] = (bf16)fc2_w[k * 384u + n];
  } else if (g < 1770624u) {
    unsigned c = g - 1769472u;
    qb[c] = (c < 384u) ? q_bias[c] : ((c < 768u) ? 0.0f : v_bias[c - 768u]);
  }
}

// ---------------------------------------------------------------------------
// Setup: CPB MLP -> 16*sigmoid bias table, region ids, head scales (1 block)
// ---------------------------------------------------------------------------
__device__ __forceinline__ float cpb_coord(int v) {
  float t = (float)v * (8.0f / 7.0f);
  float a = fabsf(t);
  float s = (t > 0.0f) ? 1.0f : ((t < 0.0f) ? -1.0f : 0.0f);
  return s * __log2f(a + 1.0f) * (1.0f / 3.0f);
}

__global__ __launch_bounds__(256) void k_setup(
    const float* logit_scale, const float* cpb_w1, const float* cpb_b1,
    const float* cpb_w2, char* ws) {
  __shared__ float table[225 * NHEAD];
  float* rpb = (float*)(ws + OFF_RPB);
  int*   reg = (int*)(ws + OFF_REGION);
  float* hsc = (float*)(ws + OFF_HSC);
  const int tid = threadIdx.x;

  if (tid < NHEAD) hsc[tid] = __expf(fminf(logit_scale[tid], 4.6051702f));

  for (int p = tid; p < 225 * NHEAD; p += 256) {
    int ci = p / NHEAD, h = p - ci * NHEAD;
    int i = ci / 15, j = ci - i * 15;
    float c0 = cpb_coord(j - 7), c1 = cpb_coord(i - 7);
    float acc = 0.0f;
    for (int u = 0; u < 512; ++u) {
      float hdn = c0 * cpb_w1[u] + c1 * cpb_w1[512 + u] + cpb_b1[u];
      hdn = fmaxf(hdn, 0.0f);
      acc += hdn * cpb_w2[u * NHEAD + h];
    }
    table[p] = acc;
  }
  __syncthreads();

  for (int p = tid; p < NHEAD * NTOK * NTOK; p += 256) {
    int h = p >> 12, ab = p & 4095;
    int a = ab >> 6, b = ab & 63;
    int dx = (a & 7) - (b & 7);
    int dy = (a >> 3) - (b >> 3);
    float v = table[((dx + 7) * 15 + (dy + 7)) * NHEAD + h];
    rpb[p] = 16.0f / (1.0f + __expf(-v));
  }

  for (int p = tid; p < 64 * 64; p += 256) {
    int w = p >> 6, t = p & 63;
    int gh = ((w >> 3) << 3) + (t >> 3);
    int gw = ((w & 7) << 3) + (t & 7);
    int rh = (gh < 56) ? 0 : ((gh < 60) ? 1 : 2);
    int rw = (gw < 56) ? 0 : ((gw < 60) ? 1 : 2);
    reg[p] = rh * 3 + rw;
  }
}

// ---------------------------------------------------------------------------
// Gather: cyclic shift + window partition + f32->bf16  (x -> XW)
// ---------------------------------------------------------------------------
__global__ __launch_bounds__(256) void k_gather(const float* X, char* ws) {
  bf16* XW = (bf16*)(ws + OFF_XW);
  const unsigned TOT = (unsigned)MTOT * 96u;
  unsigned stride = gridDim.x * 256u;
  for (unsigned ci = blockIdx.x * 256u + threadIdx.x; ci < TOT; ci += stride) {
    unsigned m = ci / 96u, c4 = ci - m * 96u;
    int col = (int)c4 * 4;
    int t = m & 63, win = m >> 6, wloc = win & 63, b = win >> 6;
    int hs = ((wloc >> 3) << 3) + (t >> 3);
    int wsx = ((wloc & 7) << 3) + (t & 7);
    int h = (hs + 4) & 63, w = (wsx + 4) & 63;
    long nl = ((long)b << 12) + (h << 6) + w;
    float4 v = *(const float4*)(X + nl * DIMC + col);
    union { bf16 h4[4]; uint2 u; } pk;
    pk.h4[0] = (bf16)v.x; pk.h4[1] = (bf16)v.y;
    pk.h4[2] = (bf16)v.z; pk.h4[3] = (bf16)v.w;
    *(uint2*)(XW + (long)m * DIMC + col) = pk.u;
  }
}

// ---------------------------------------------------------------------------
// QKV GEMM: 64 rows x 384-col slab. Q/K stored row-major; V stored transposed
// per (window, head) as VT[wh][d][t] so attention reads K-contiguous frags.
// ---------------------------------------------------------------------------
__global__ __launch_bounds__(256) void k_qkv(char* ws) {
  const bf16* XW = (const bf16*)(ws + OFF_XW);
  const bf16* Wq = (const bf16*)(ws + OFF_WQKV);    // transposed [1152][384]
  const float* qb = (const float*)(ws + OFF_QKVB);
  bf16* QKV = (bf16*)(ws + OFF_QKV);
  bf16* VT  = (bf16*)(ws + OFF_VT);

  __shared__ bf16 bl[384 * BSTR];
  const unsigned bl_off = (unsigned)(uintptr_t)(void*)bl;
  const int tid = threadIdx.x, lane = tid & 31, wave = tid >> 5;
  const int wm = wave & 3, wn = wave >> 2;
  const int m0 = blockIdx.x * 64;
  const int n0s = blockIdx.y * 384;

  v8f acc[12];
#pragma unroll
  for (int i = 0; i < 12; ++i) acc[i] = zero8();

  for (int ks = 0; ks < 12; ++ks) {
    __syncthreads();
    stage_async<384>(Wq, 384, n0s, ks * 32, bl_off, tid);
    if (ks < 11)
      __builtin_prefetch(XW + (long)(m0 + wm * 16 + (lane & 15)) * DIMC +
                             (ks + 1) * 32, 0, 1);
    async_wait0();
    __syncthreads();
    v16bf a = afrag(XW, DIMC, m0 + wm * 16, ks * 32, lane);
#pragma unroll
    for (int nt = 0; nt < 12; ++nt) {
      v16bf b = bfrag(bl, BSTR, wn * 192 + nt * 16, 0, lane);
      acc[nt] = wmma_bf16(a, b, acc[nt]);
    }
  }

  const int hi8 = (lane >> 4) << 3, lo = lane & 15;
#pragma unroll
  for (int nt = 0; nt < 12; ++nt) {
    int cc = wn * 192 + nt * 16 + lo;           // col within slab
    float bia = qb[n0s + cc];
    if (n0s == 768) {                           // V slab -> transposed layout
      int head = cc >> 5, d = cc & 31;
#pragma unroll
      for (int r = 0; r < 8; ++r) {
        int row = m0 + wm * 16 + r + hi8;
        int win = row >> 6, t = row & 63;
        VT[((long)win * NHEAD + head) * 2048 + d * 64 + t] =
            (bf16)(acc[nt][r] + bia);
      }
    } else {
#pragma unroll
      for (int r = 0; r < 8; ++r) {
        int row = m0 + wm * 16 + r + hi8;
        QKV[(long)row * LDQKV + n0s + cc] = (bf16)(acc[nt][r] + bia);
      }
    }
  }
}

// ---------------------------------------------------------------------------
// Fused cosine attention: one wave = one (window, head). 3 waves / block.
// ---------------------------------------------------------------------------
#define AW 3
#define PW_F 4224   // per-wave floats: S 4096 + qinv 64 + kinv 64

__global__ __launch_bounds__(96) void k_attn(char* ws) {
  const bf16* QKV = (const bf16*)(ws + OFF_QKV);
  const bf16* VT = (const bf16*)(ws + OFF_VT);
  const float* RPB = (const float*)(ws + OFF_RPB);
  const int* REG = (const int*)(ws + OFF_REGION);
  const float* HSC = (const float*)(ws + OFF_HSC);
  bf16* OUT = (bf16*)(ws + OFF_XW);   // alias: XW reused as ATTOUT

  __shared__ float smem[AW * PW_F];
  const int tid = threadIdx.x, lane = tid & 31, wave = tid >> 5;
  float* Sf = smem + wave * PW_F;
  float* qinv = Sf + 4096;
  float* kinv = qinv + 64;

  const int wid = blockIdx.x * AW + wave;
  const int win = wid / NHEAD, head = wid - win * NHEAD;
  const int wloc = win & 63;
  const bf16* Qp = QKV + (long)win * 64 * LDQKV + head * HDIM;
  const bf16* Kp = Qp + DIMC;
  const bf16* Vh = VT + ((long)win * NHEAD + head) * 2048;  // [d][t]

  // ---- phase 1: inverse norms ----
#pragma unroll
  for (int t2 = 0; t2 < 2; ++t2) {
    int t = lane + t2 * 32;
    float sq = 0.0f, sk = 0.0f;
#pragma unroll
    for (int c = 0; c < 4; ++c) {
      v8bf q = *(const v8bf*)(Qp + (long)t * LDQKV + c * 8);
      v8bf k = *(const v8bf*)(Kp + (long)t * LDQKV + c * 8);
#pragma unroll
      for (int e = 0; e < 8; ++e) {
        float qf = (float)q[e], kf = (float)k[e];
        sq += qf * qf; sk += kf * kf;
      }
    }
    qinv[t] = rsqrtf(fmaxf(sq, 1e-24f));
    kinv[t] = rsqrtf(fmaxf(sk, 1e-24f));
  }
  __syncthreads();

  // ---- phase 2: S = scale*(q.k)/(|q||k|) + rpb + mask (1 WMMA per tile) ----
  const float hs = HSC[head];
  const int hi8 = (lane >> 4) << 3, lo = lane & 15;
#pragma unroll
  for (int mi = 0; mi < 4; ++mi) {
    const bf16* ap = Qp + (long)(mi * 16 + lo) * LDQKV + ((lane >> 4) << 3);
    v16bf aq = ld_pair16(ap, ap + 16);
    float qv[8]; int ri[8];
#pragma unroll
    for (int r = 0; r < 8; ++r) {
      int row = mi * 16 + r + hi8;
      qv[r] = qinv[row];
      ri[r] = REG[wloc * 64 + row];
    }
#pragma unroll
    for (int ni = 0; ni < 4; ++ni) {
      const bf16* bp = Kp + (long)(ni * 16 + lo) * LDQKV + ((lane >> 4) << 3);
      v16bf bk = ld_pair16(bp, bp + 16);
      v8f s = wmma_bf16(aq, bk, zero8());
      int col = ni * 16 + lo;
      float kv = kinv[col];
      int rj = REG[wloc * 64 + col];
      const float* rp = RPB + head * 4096 + col;
#pragma unroll
      for (int r = 0; r < 8; ++r) {
        int row = mi * 16 + r + hi8;
        float v = s[r] * hs * qv[r] * kv + rp[row * 64] +
                  ((ri[r] == rj) ? 0.0f : -100.0f);
        Sf[row * 64 + col] = v;
      }
    }
  }
  __syncthreads();

  // ---- phase 3: softmax rows; overwrite S with bf16 P in place ----
#pragma unroll
  for (int t2 = 0; t2 < 2; ++t2) {
    int r = lane + t2 * 32;
    float* Sr = Sf + r * 64;
    float mx = -1e30f;
    for (int c = 0; c < 64; ++c) mx = fmaxf(mx, Sr[c]);
    float sm = 0.0f;
    for (int c = 0; c < 64; ++c) sm += __expf(Sr[c] - mx);
    float inv = 1.0f / sm;
    bf16* Pr = (bf16*)Sr;
    for (int c = 0; c < 64; ++c) Pr[c] = (bf16)(__expf(Sr[c] - mx) * inv);
  }
  __syncthreads();

  // ---- phase 4: O = P @ V (V fragments K-contiguous from global VT) ----
  const bf16* P = (const bf16*)Sf;   // row stride 128 bf16
#pragma unroll
  for (int mi = 0; mi < 4; ++mi) {
    v8f o0 = zero8(), o1 = zero8();
#pragma unroll
    for (int kk = 0; kk < 2; ++kk) {
      const bf16* ap = P + (long)(mi * 16 + lo) * 128 + kk * 32 + ((lane >> 4) << 3);
      v16bf a = ld_pair16(ap, ap + 16);
      const bf16* b0 = Vh + (long)lo * 64 + kk * 32 + ((lane >> 4) << 3);
      o0 = wmma_bf16(a, ld_pair16(b0, b0 + 16), o0);
      const bf16* b1 = Vh + (long)(16 + lo) * 64 + kk * 32 + ((lane >> 4) << 3);
      o1 = wmma_bf16(a, ld_pair16(b1, b1 + 16), o1);
    }
#pragma unroll
    for (int r = 0; r < 8; ++r) {
      int row = mi * 16 + r + hi8;
      bf16* op = OUT + (long)(win * 64 + row) * DIMC + head * HDIM;
      op[lo] = (bf16)o0[r];
      op[16 + lo] = (bf16)o1[r];
    }
  }
}

// ---------------------------------------------------------------------------
// Proj GEMM + window-reverse + unshift + residual(x) + LayerNorm(norm1)
// ---------------------------------------------------------------------------
__global__ __launch_bounds__(256) void k_proj(const float* X, const float* pb,
                                              const float* g1, const float* b1,
                                              char* ws) {
  const bf16* A = (const bf16*)(ws + OFF_XW);       // ATTOUT
  const bf16* Wp = (const bf16*)(ws + OFF_WPROJ);   // transposed [384][384]
  float* XMf = (float*)(ws + OFF_XMIDF);
  bf16* XMb = (bf16*)(ws + OFF_XMIDB);

  __shared__ bf16 bl[384 * BSTR];
  __shared__ float stats[64][2][2];
  const unsigned bl_off = (unsigned)(uintptr_t)(void*)bl;
  const int tid = threadIdx.x, lane = tid & 31, wave = tid >> 5;
  const int wm = wave & 3, wn = wave >> 2;
  const int m0 = blockIdx.x * 64;
  const int hi8 = (lane >> 4) << 3, lo = lane & 15;

  v8f acc[12];
#pragma unroll
  for (int i = 0; i < 12; ++i) acc[i] = zero8();

  for (int ks = 0; ks < 12; ++ks) {
    __syncthreads();
    stage_async<384>(Wp, 384, 0, ks * 32, bl_off, tid);
    if (ks < 11)
      __builtin_prefetch(A + (long)(m0 + wm * 16 + (lane & 15)) * DIMC +
                             (ks + 1) * 32, 0, 1);
    async_wait0();
    __syncthreads();
    v16bf a = afrag(A, DIMC, m0 + wm * 16, ks * 32, lane);
#pragma unroll
    for (int nt = 0; nt < 12; ++nt) {
      v16bf b = bfrag(bl, BSTR, wn * 192 + nt * 16, 0, lane);
      acc[nt] = wmma_bf16(a, b, acc[nt]);
    }
  }

#pragma unroll
  for (int nt = 0; nt < 12; ++nt) {
    float bia = pb[wn * 192 + nt * 16 + lo];
#pragma unroll
    for (int r = 0; r < 8; ++r) acc[nt][r] += bia;
  }
#pragma unroll
  for (int r = 0; r < 8; ++r) {
    float s = 0.0f, q = 0.0f;
#pragma unroll
    for (int nt = 0; nt < 12; ++nt) { float v = acc[nt][r]; s += v; q += v * v; }
#pragma unroll
    for (int off = 1; off < 16; off <<= 1) {
      s += __shfl_xor(s, off, 16);
      q += __shfl_xor(q, off, 16);
    }
    if (lo == 0) {
      stats[wm * 16 + r + hi8][0][wn] = s;
      stats[wm * 16 + r + hi8][1][wn] = q;
    }
  }
  __syncthreads();

#pragma unroll
  for (int r = 0; r < 8; ++r) {
    int rowl = wm * 16 + r + hi8;
    float mean = (stats[rowl][0][0] + stats[rowl][0][1]) * (1.0f / DIMC);
    float var = (stats[rowl][1][0] + stats[rowl][1][1]) * (1.0f / DIMC) - mean * mean;
    float rstd = rsqrtf(var + 1e-6f);
    int m = m0 + rowl;
    int t = m & 63, win = m >> 6, wloc = win & 63, b = win >> 6;
    int hs = ((wloc >> 3) << 3) + (t >> 3);
    int wsx = ((wloc & 7) << 3) + (t & 7);
    int h = (hs + 4) & 63, w = (wsx + 4) & 63;
    long nl = ((long)b << 12) + (h << 6) + w;
#pragma unroll
    for (int nt = 0; nt < 12; ++nt) {
      int col = wn * 192 + nt * 16 + lo;
      float o = X[nl * DIMC + col] + (acc[nt][r] - mean) * rstd * g1[col] + b1[col];
      XMf[nl * DIMC + col] = o;
      XMb[nl * DIMC + col] = (bf16)o;
    }
  }
}

// ---------------------------------------------------------------------------
// Fused MLP: y = gelu(xmid@fc1+b1)@fc2+b2 ; out = xmid + LN2(y)
// Hidden processed in 12 chunks of 128 via LDS; 64x384 fc2 acc stays in VGPRs.
// ---------------------------------------------------------------------------
__global__ __launch_bounds__(256) void k_mlp(const float* f1b, const float* f2b,
                                             const float* g2, const float* b2,
                                             float* OUTF, char* ws) {
  const bf16* A = (const bf16*)(ws + OFF_XMIDB);
  const float* XMf = (const float*)(ws + OFF_XMIDF);
  const bf16* W1 = (const bf16*)(ws + OFF_WFC1);    // transposed [1536][384]
  const bf16* W2 = (const bf16*)(ws + OFF_WFC2);    // transposed [384][1536]

  __shared__ bf16 bl[384 * BSTR];
  __shared__ bf16 hl[64 * 136];
  __shared__ float stats[64][2][2];
  const unsigned bl_off = (unsigned)(uintptr_t)(void*)bl;
  const int tid = threadIdx.x, lane = tid & 31, wave = tid >> 5;
  const int wm = wave & 3, wn = wave >> 2;
  const int m0 = blockIdx.x * 64;
  const int hi8 = (lane >> 4) << 3, lo = lane & 15;

  v8f acc2[12];
#pragma unroll
  for (int i = 0; i < 12; ++i) acc2[i] = zero8();

  for (int ch = 0; ch < 12; ++ch) {
    v8f acc1[4];
#pragma unroll
    for (int i = 0; i < 4; ++i) acc1[i] = zero8();
    for (int ks = 0; ks < 12; ++ks) {
      __syncthreads();
      stage_async<128>(W1, 384, ch * 128, ks * 32, bl_off, tid);
      async_wait0();
      __syncthreads();
      v16bf a = afrag(A, DIMC, m0 + wm * 16, ks * 32, lane);
#pragma unroll
      for (int nt = 0; nt < 4; ++nt) {
        v16bf b = bfrag(bl, BSTR, wn * 64 + nt * 16, 0, lane);
        acc1[nt] = wmma_bf16(a, b, acc1[nt]);
      }
    }
    __syncthreads();
#pragma unroll
    for (int nt = 0; nt < 4; ++nt) {
      int col = wn * 64 + nt * 16 + lo;
      float bia = f1b[ch * 128 + col];
#pragma unroll
      for (int r = 0; r < 8; ++r) {
        float x = acc1[nt][r] + bia;
        float u = 0.7978845608f * (x + 0.044715f * x * x * x);
        float g = 0.5f * x * (1.0f + tanhf(u));
        hl[(wm * 16 + r + hi8) * 136 + col] = (bf16)g;
      }
    }
    for (int ks = 0; ks < 4; ++ks) {
      __syncthreads();
      stage_async<384>(W2, 1536, 0, ch * 128 + ks * 32, bl_off, tid);
      async_wait0();
      __syncthreads();
      v16bf a = afrag(hl, 136, wm * 16, ks * 32, lane);
#pragma unroll
      for (int nt = 0; nt < 12; ++nt) {
        v16bf b = bfrag(bl, BSTR, wn * 192 + nt * 16, 0, lane);
        acc2[nt] = wmma_bf16(a, b, acc2[nt]);
      }
    }
  }

#pragma unroll
  for (int nt = 0; nt < 12; ++nt) {
    float bia = f2b[wn * 192 + nt * 16 + lo];
#pragma unroll
    for (int r = 0; r < 8; ++r) acc2[nt][r] += bia;
  }
#pragma unroll
  for (int r = 0; r < 8; ++r) {
    float s = 0.0f, q = 0.0f;
#pragma unroll
    for (int nt = 0; nt < 12; ++nt) { float v = acc2[nt][r]; s += v; q += v * v; }
#pragma unroll
    for (int off = 1; off < 16; off <<= 1) {
      s += __shfl_xor(s, off, 16);
      q += __shfl_xor(q, off, 16);
    }
    if (lo == 0) {
      stats[wm * 16 + r + hi8][0][wn] = s;
      stats[wm * 16 + r + hi8][1][wn] = q;
    }
  }
  __syncthreads();
#pragma unroll
  for (int r = 0; r < 8; ++r) {
    int rowl = wm * 16 + r + hi8;
    float mean = (stats[rowl][0][0] + stats[rowl][0][1]) * (1.0f / DIMC);
    float var = (stats[rowl][1][0] + stats[rowl][1][1]) * (1.0f / DIMC) - mean * mean;
    float rstd = rsqrtf(var + 1e-6f);
    long m = m0 + rowl;
#pragma unroll
    for (int nt = 0; nt < 12; ++nt) {
      int col = wn * 192 + nt * 16 + lo;
      OUTF[m * DIMC + col] =
          XMf[m * DIMC + col] + (acc2[nt][r] - mean) * rstd * g2[col] + b2[col];
    }
  }
}

// ---------------------------------------------------------------------------
extern "C" void kernel_launch(void* const* d_in, const int* in_sizes, int n_in,
                              void* d_out, int out_size, void* d_ws,
                              size_t ws_size, hipStream_t stream) {
  (void)in_sizes; (void)n_in; (void)out_size; (void)ws_size;
  const float* x      = (const float*)d_in[0];
  const float* qkv_w  = (const float*)d_in[1];
  const float* q_bias = (const float*)d_in[2];
  const float* v_bias = (const float*)d_in[3];
  const float* lscale = (const float*)d_in[4];
  const float* cpb_w1 = (const float*)d_in[5];
  const float* cpb_b1 = (const float*)d_in[6];
  const float* cpb_w2 = (const float*)d_in[7];
  const float* proj_w = (const float*)d_in[8];
  const float* proj_b = (const float*)d_in[9];
  const float* n1s    = (const float*)d_in[10];
  const float* n1b    = (const float*)d_in[11];
  const float* fc1_w  = (const float*)d_in[12];
  const float* fc1_b  = (const float*)d_in[13];
  const float* fc2_w  = (const float*)d_in[14];
  const float* fc2_b  = (const float*)d_in[15];
  const float* n2s    = (const float*)d_in[16];
  const float* n2b    = (const float*)d_in[17];
  char* ws = (char*)d_ws;

  k_convert<<<6917, 256, 0, stream>>>(qkv_w, proj_w, fc1_w, fc2_w, q_bias,
                                      v_bias, ws);
  k_setup<<<1, 256, 0, stream>>>(lscale, cpb_w1, cpb_b1, cpb_w2, ws);
  k_gather<<<12288, 256, 0, stream>>>(x, ws);
  k_qkv<<<dim3(MTOT / 64, 3), 256, 0, stream>>>(ws);
  k_attn<<<(MTOT / 64 * NHEAD) / AW, 96, 0, stream>>>(ws);
  k_proj<<<MTOT / 64, 256, 0, stream>>>(x, proj_b, n1s, n1b, ws);
  k_mlp<<<MTOT / 64, 256, 0, stream>>>(fc1_b, fc2_b, n2s, n2b, (float*)d_out,
                                       ws);
}
